// _TELayerNormLinear_73375221285154
// MI455X (gfx1250) — matrix operand order repaired and checked by reading, not compiled, not run on hardware
//
#include <hip/hip_runtime.h>
#include <hip/hip_bf16.h>
#include <stdint.h>

#define D_IN   4096
#define D_OUT  12288
#define M_TOT  8192   // B*S = 4*2048

typedef __attribute__((ext_vector_type(16))) __bf16 v16bf;
typedef __attribute__((ext_vector_type(8)))  __bf16 v8bf;
typedef __attribute__((ext_vector_type(8)))  float  v8f;

static __device__ __forceinline__ unsigned int pack_bf16x2(float a, float b) {
    union { float f; uint32_t u; } ca{a}, cb{b};
    uint32_t ua = ca.u + (0x7FFFu + ((ca.u >> 16) & 1u));
    uint32_t ub = cb.u + (0x7FFFu + ((cb.u >> 16) & 1u));
    return (ua >> 16) | (ub & 0xFFFF0000u);
}

// ---------------------------------------------------------------------------
// Kernel 1: LayerNorm one row per block (256 threads, 16 elems/thread),
// write normalized activations as bf16 (row-major M x K).
// ---------------------------------------------------------------------------
__global__ void __launch_bounds__(256) ln_bf16_kernel(
    const float* __restrict__ x,
    const float* __restrict__ g,
    const float* __restrict__ b,
    unsigned short* __restrict__ out) {

    const int row = blockIdx.x;
    const float4* xr = (const float4*)(x + (size_t)row * D_IN);
    const float4* gv = (const float4*)g;
    const float4* bv = (const float4*)b;

    float4 v[4];
    float s = 0.f, ss = 0.f;
#pragma unroll
    for (int i = 0; i < 4; ++i) {
        v[i] = xr[threadIdx.x + i * 256];
        s  += v[i].x + v[i].y + v[i].z + v[i].w;
        ss += v[i].x * v[i].x + v[i].y * v[i].y + v[i].z * v[i].z + v[i].w * v[i].w;
    }
    // wave32 reduction
#pragma unroll
    for (int off = 16; off > 0; off >>= 1) {
        s  += __shfl_xor(s,  off, 32);
        ss += __shfl_xor(ss, off, 32);
    }
    __shared__ float s_sum[8];
    __shared__ float s_ssq[8];
    const int wave = threadIdx.x >> 5;
    if ((threadIdx.x & 31) == 0) { s_sum[wave] = s; s_ssq[wave] = ss; }
    __syncthreads();
    float ts = 0.f, tss = 0.f;
#pragma unroll
    for (int i = 0; i < 8; ++i) { ts += s_sum[i]; tss += s_ssq[i]; }

    const float inv_n = 1.0f / (float)D_IN;
    const float mean  = ts * inv_n;
    const float var   = tss * inv_n - mean * mean;
    const float rstd  = rsqrtf(var + 1e-5f);

    uint2* o2 = (uint2*)(out + (size_t)row * D_IN);
#pragma unroll
    for (int i = 0; i < 4; ++i) {
        const int idx4 = threadIdx.x + i * 256;
        float4 gw = gv[idx4];
        float4 bb = bv[idx4];
        float y0 = (v[i].x - mean) * rstd * gw.x + bb.x;
        float y1 = (v[i].y - mean) * rstd * gw.y + bb.y;
        float y2 = (v[i].z - mean) * rstd * gw.z + bb.z;
        float y3 = (v[i].w - mean) * rstd * gw.w + bb.w;
        uint2 p;
        p.x = pack_bf16x2(y0, y1);
        p.y = pack_bf16x2(y2, y3);
        o2[idx4] = p;
    }
}

// ---------------------------------------------------------------------------
// Kernel 2: f32 -> bf16 weight conversion (grid-stride, float4 loads)
// ---------------------------------------------------------------------------
__global__ void __launch_bounds__(256) cvt_bf16_kernel(
    const float* __restrict__ w,
    unsigned short* __restrict__ wb,
    long n4) {
    const float4* w4 = (const float4*)w;
    uint2* o2 = (uint2*)wb;
    for (long i = (long)blockIdx.x * blockDim.x + threadIdx.x; i < n4;
         i += (long)gridDim.x * blockDim.x) {
        float4 f = w4[i];
        uint2 p;
        p.x = pack_bf16x2(f.x, f.y);
        p.y = pack_bf16x2(f.z, f.w);
        o2[i] = p;
    }
}

// ---------------------------------------------------------------------------
// Kernel 3: bf16 WMMA GEMM  C[M,N] = A[M,K] * W[N,K]^T + bias
// Block: 256 threads = 8 waves (2x4). Wave tile 64x64 -> block tile 128x256.
// K loop step 32 using v_wmma_f32_16x16x32_bf16.
// ---------------------------------------------------------------------------
__global__ void __launch_bounds__(256) wmma_gemm_kernel(
    const unsigned short* __restrict__ Araw,   // M x K bf16
    const unsigned short* __restrict__ Wraw,   // N x K bf16
    const float* __restrict__ bias,            // N
    float* __restrict__ C) {                   // M x N f32

    const __bf16* A = (const __bf16*)Araw;
    const __bf16* W = (const __bf16*)Wraw;

    const int lane   = threadIdx.x & 31;
    const int wave   = threadIdx.x >> 5;
    const int wRow   = wave >> 2;            // 0..1
    const int wCol   = wave & 3;             // 0..3
    const int rowBase = blockIdx.y * 128 + wRow * 64;
    const int colBase = blockIdx.x * 256 + wCol * 64;
    const int lh     = lane & 15;
    const bool hi    = lane >= 16;

    v8f acc[4][4];
#pragma unroll
    for (int i = 0; i < 4; ++i)
#pragma unroll
        for (int j = 0; j < 4; ++j)
            acc[i][j] = v8f{0.f, 0.f, 0.f, 0.f, 0.f, 0.f, 0.f, 0.f};

    // A fragment (16x32 bf16, MxK): lanes 0-15 hold K 0..7 & 16..23 of row M=lh;
    // lanes 16-31 hold K 8..15 & 24..31.  -> base offset +8 for hi lanes,
    // two 16B loads at +0 and +16 elements.
    const __bf16* aPtr[4];
#pragma unroll
    for (int i = 0; i < 4; ++i)
        aPtr[i] = A + (size_t)(rowBase + i * 16 + lh) * D_IN + (hi ? 8 : 0);

    // B fragment (32x16 bf16, KxN): lane n=lh holds K 0..7 & 16..23 for lanes 0-15,
    // K 8..15 & 24..31 for lanes 16-31; row n of W is
    // contiguous in K.  -> base +8 for hi lanes, two 16B loads at +0,+16.
    const __bf16* bPtr[4];
#pragma unroll
    for (int j = 0; j < 4; ++j)
        bPtr[j] = W + (size_t)(colBase + j * 16 + lh) * D_IN + (hi ? 8 : 0);

    for (int k = 0; k < D_IN; k += 32) {
        v16bf afrag[4], bfrag[4];
#pragma unroll
        for (int i = 0; i < 4; ++i) {
            union { v16bf v; v8bf h[2]; } u;
            u.h[0] = *(const v8bf*)(aPtr[i] + k);
            u.h[1] = *(const v8bf*)(aPtr[i] + k + 16);
            afrag[i] = u.v;
        }
#pragma unroll
        for (int j = 0; j < 4; ++j) {
            union { v16bf v; v8bf h[2]; } u;
            u.h[0] = *(const v8bf*)(bPtr[j] + k);
            u.h[1] = *(const v8bf*)(bPtr[j] + k + 16);
            bfrag[j] = u.v;
        }
#pragma unroll
        for (int i = 0; i < 4; ++i)
#pragma unroll
            for (int j = 0; j < 4; ++j)
                acc[i][j] = __builtin_amdgcn_wmma_f32_16x16x32_bf16(
                    false, afrag[i], false, bfrag[j],
                    (short)0, acc[i][j], false, false);
    }

    // Epilogue: C/D layout -> VGPR r: lanes 0-15 row r, lanes 16-31 row r+8;
    // column = lane&15 within the 16-wide tile.
#pragma unroll
    for (int i = 0; i < 4; ++i) {
#pragma unroll
        for (int j = 0; j < 4; ++j) {
            const int col = colBase + j * 16 + lh;
            const float bvv = bias[col];
            const int r0 = rowBase + i * 16 + (hi ? 8 : 0);
#pragma unroll
            for (int v = 0; v < 8; ++v) {
                C[(size_t)(r0 + v) * D_OUT + col] = acc[i][j][v] + bvv;
            }
        }
    }
}

extern "C" void kernel_launch(void* const* d_in, const int* in_sizes, int n_in,
                              void* d_out, int out_size, void* d_ws, size_t ws_size,
                              hipStream_t stream) {
    const float* x      = (const float*)d_in[0];
    const float* weight = (const float*)d_in[1];
    const float* bias   = (const float*)d_in[2];
    const float* ln_w   = (const float*)d_in[3];
    const float* ln_b   = (const float*)d_in[4];
    float* out = (float*)d_out;

    unsigned short* aN = (unsigned short*)d_ws;                               // M x K bf16
    unsigned short* wB = (unsigned short*)((char*)d_ws +
                         (size_t)M_TOT * D_IN * sizeof(unsigned short));      // N x K bf16

    ln_bf16_kernel<<<M_TOT, 256, 0, stream>>>(x, ln_w, ln_b, aN);

    const long n4 = (long)D_OUT * D_IN / 4;
    cvt_bf16_kernel<<<4096, 256, 0, stream>>>(weight, wB, n4);

    dim3 grid(D_OUT / 256, M_TOT / 128);
    wmma_gemm_kernel<<<grid, 256, 0, stream>>>(aN, wB, bias, out);
}
